// Spiking_Net_31482110280234
// MI455X (gfx1250) — matrix operand order repaired and checked by reading, not compiled, not run on hardware
//
#include <hip/hip_runtime.h>

// ---------------- model constants ----------------
#define BETA    0.9f
#define THRESH  1.0f
#define T_STEPS 25
#define B_ROWS  256
#define L0 2048
#define L1 4096
#define L2 4096
#define L3 1024

// ---------------- vector types -------------------
typedef __attribute__((ext_vector_type(16))) __bf16 v16bf;
typedef __attribute__((ext_vector_type(8)))  __bf16 v8bf;
typedef __attribute__((ext_vector_type(8)))  float  v8f;

union Frag { v16bf v; v8bf h[2]; };

// ---------------- helpers ------------------------
__device__ __forceinline__ unsigned short f32_to_bf16_bits(float f) {
    unsigned int u = __float_as_uint(f);
    unsigned int r = (u + 0x7FFFu + ((u >> 16) & 1u)) >> 16;   // round-to-nearest-even
    return (unsigned short)r;
}

__device__ __forceinline__ unsigned long long mix64(unsigned long long x) {
    x += 0x9E3779B97F4A7C15ull;
    x = (x ^ (x >> 30)) * 0xBF58476D1CE4E5B9ull;
    x = (x ^ (x >> 27)) * 0x94D049BB133111EBull;
    return x ^ (x >> 31);
}

// ---------------- utility kernels ----------------
__global__ void cvt_bf16_kernel(const float* __restrict__ src,
                                unsigned short* __restrict__ dst, int n) {
    int i = blockIdx.x * blockDim.x + threadIdx.x;
    if (i < n) dst[i] = f32_to_bf16_bits(src[i]);
}

__global__ void zero_f32_kernel(float* __restrict__ p, int n) {
    int i = blockIdx.x * blockDim.x + threadIdx.x;
    if (i < n) p[i] = 0.0f;
}

// Bernoulli rate-coded spike generation -> bf16 {0,1}, shape (T, B, L0)
__global__ void spikegen_kernel(const float* __restrict__ data,
                                unsigned short* __restrict__ X, int total) {
    int i = blockIdx.x * blockDim.x + threadIdx.x;
    if (i >= total) return;
    int bi = i % (B_ROWS * L0);                       // broadcast p over T
    float p = data[bi];
    p = fminf(fmaxf(p, 0.0f), 1.0f);
    unsigned long long h = mix64((unsigned long long)i ^ 0xD6E8FEB86659FD93ull);
    float u = (float)(h >> 40) * (1.0f / 16777216.0f);  // 24-bit uniform [0,1)
    X[i] = (u < p) ? (unsigned short)0x3F80 : (unsigned short)0x0000;
}

// ---------------- fused WMMA GEMM + LIF ----------
// cur(MxN) = A(MxK) @ W(NxK)^T ; then in-epilogue LIF:
//   mem = BETA*mem + cur + bias[n]; spk = (mem-THRESH > 0); mem -= spk
// A,W bf16 (as u16); mem f32 in-place; spk bf16; optional f32 spike record.
// Block: 256 threads = 8 waves; tile BM x BN x BK = 64x128x64
// Waves arranged 4(M) x 2(N); each wave -> 16x64 output (4 accumulators)
// Staging: GLOBAL_LOAD_ASYNC_TO_LDS_B128 (ASYNCcnt), double buffered.
#define BM 64
#define BN 128
#define BK 64
#define LDK (BK + 8)   // padded row stride (bf16 elems); 144B keeps 16B align

__global__ __launch_bounds__(256)
void gemm_lif_bf16_wmma_kernel(const unsigned short* __restrict__ A,
                               const unsigned short* __restrict__ W,
                               const float* __restrict__ bias,
                               float* __restrict__ mem,
                               unsigned short* __restrict__ spk,
                               float* __restrict__ out_f32,   // may be null
                               int M, int N, int K) {
    __shared__ __bf16 As[2][BM][LDK];
    __shared__ __bf16 Bs[2][BN][LDK];

    const int tid   = threadIdx.x;
    const int lane  = tid & 31;
    const int wave  = tid >> 5;
    const int waveM = wave & 3;    // 0..3 -> 16-row slice
    const int waveN = wave >> 2;   // 0..1 -> 64-col slice
    const int m0 = blockIdx.y * BM;
    const int n0 = blockIdx.x * BN;
    const int hi16 = (lane >> 4);  // 0 or 1

    // Raw LDS byte offsets (generic-pointer low 32 bits == LDS address)
    const unsigned asBase = (unsigned)(size_t)(&As[0][0][0]);
    const unsigned bsBase = (unsigned)(size_t)(&Bs[0][0][0]);

    // Per-thread staging slots: A = 512 slots (64 rows x 8 segs),
    //                           B = 1024 slots (128 rows x 8 segs). 16B each.
    const int rowA = tid >> 3;         // 0..31 (+32k for further slots)
    const int segA = tid & 7;

    v8f acc[4] = {v8f{}, v8f{}, v8f{}, v8f{}};

    const int nK = K / BK;

    // issue one tile's async loads into buffer s (no VGPR round-trip)
    auto loadTileAsync = [&](int s, int kbase) {
#pragma unroll
        for (int it = 0; it < 2; ++it) {           // A tile: 64x64 bf16
            const int row = rowA + it * 32;
            const unsigned lds = asBase +
                (unsigned)(((s * BM + row) * LDK + segA * 8) * 2);
            const unsigned long long g = (unsigned long long)
                (A + (size_t)(m0 + row) * K + kbase + segA * 8);
            asm volatile("global_load_async_to_lds_b128 %0, %1, off"
                         :: "v"(lds), "v"(g) : "memory");
        }
#pragma unroll
        for (int it = 0; it < 4; ++it) {           // B tile: 128x64 bf16
            const int row = rowA + it * 32;
            const unsigned lds = bsBase +
                (unsigned)(((s * BN + row) * LDK + segA * 8) * 2);
            const unsigned long long g = (unsigned long long)
                (W + (size_t)(n0 + row) * K + kbase + segA * 8);
            asm volatile("global_load_async_to_lds_b128 %0, %1, off"
                         :: "v"(lds), "v"(g) : "memory");
        }
    };

    loadTileAsync(0, 0);
    asm volatile("s_wait_asynccnt 0" ::: "memory");
    __syncthreads();

    for (int kb = 0; kb < nK; ++kb) {
        const int s = kb & 1;
        if (kb + 1 < nK) loadTileAsync(s ^ 1, (kb + 1) * BK);
        if (kb + 2 < nK) {  // pull tile-after-next into L2 (global_prefetch_b8)
            __builtin_prefetch(A + (size_t)(m0 + rowA) * K + (kb + 2) * BK + segA * 8, 0, 1);
            __builtin_prefetch(W + (size_t)(n0 + rowA) * K + (kb + 2) * BK + segA * 8, 0, 1);
        }

        // ---- load ALL fragments for this tile first (lets the scheduler
        //      overlap ds_loads with WMMAs instead of dscnt-0 per WMMA) ----
        Frag a[2];
        Frag b[2][4];
        const int arow = waveM * 16 + (lane & 15);
        const int bn   = waveN * 64 + (lane & 15);
#pragma unroll
        for (int kk2 = 0; kk2 < 2; ++kk2) {
            const int kk  = kk2 * 32;
            // A fragment (16x32): lanes 0-15 row M=lane, K {0-7,16-23};
            //                     lanes 16-31 same rows, K {8-15,24-31}
            const int ak0 = kk + (hi16 ? 8 : 0);
            a[kk2].h[0] = *(const v8bf*)(&As[s][arow][ak0]);
            a[kk2].h[1] = *(const v8bf*)(&As[s][arow][ak0 + 16]);
            // B fragments (32x16): lane col N=lane%16; lanes 0-15 K {0-7,16-23},
            //                      lanes 16-31 K {8-15,24-31}
            const int bk0 = kk + (hi16 ? 8 : 0);
#pragma unroll
            for (int nn = 0; nn < 4; ++nn) {
                b[kk2][nn].h[0] = *(const v8bf*)(&Bs[s][bn + nn * 16][bk0]);
                b[kk2][nn].h[1] = *(const v8bf*)(&Bs[s][bn + nn * 16][bk0 + 16]);
            }
        }
#pragma unroll
        for (int kk2 = 0; kk2 < 2; ++kk2) {
#pragma unroll
            for (int nn = 0; nn < 4; ++nn) {
                acc[nn] = __builtin_amdgcn_wmma_f32_16x16x32_bf16(
                    false, a[kk2].v, false, b[kk2][nn].v,
                    (short)0, acc[nn], false, false);
            }
        }

        if (kb + 1 < nK) asm volatile("s_wait_asynccnt 0" ::: "memory");
        __syncthreads();
    }

    // ---- fused LIF epilogue ----
    // C/D layout: VGPR v -> M = v + 8*(lane>=16); N = lane%16
#pragma unroll
    for (int nn = 0; nn < 4; ++nn) {
        const int n = n0 + waveN * 64 + nn * 16 + (lane & 15);
        const float bn_bias = bias[n];
#pragma unroll
        for (int v = 0; v < 8; ++v) {
            const int m = m0 + waveM * 16 + v + (hi16 << 3);
            const size_t idx = (size_t)m * N + n;
            float mm = BETA * mem[idx] + acc[nn][v] + bn_bias;
            const float sp = ((mm - THRESH) > 0.0f) ? 1.0f : 0.0f;
            mm -= sp * THRESH;
            mem[idx] = mm;
            spk[idx] = (sp > 0.0f) ? (unsigned short)0x3F80 : (unsigned short)0x0000;
            if (out_f32) out_f32[idx] = sp;
        }
    }
}

// ---------------- host-side orchestration --------
extern "C" void kernel_launch(void* const* d_in, const int* in_sizes, int n_in,
                              void* d_out, int out_size, void* d_ws, size_t ws_size,
                              hipStream_t stream) {
    const float* data = (const float*)d_in[0];
    const float* W1   = (const float*)d_in[1];
    const float* b1   = (const float*)d_in[2];
    const float* W2   = (const float*)d_in[3];
    const float* b2   = (const float*)d_in[4];
    const float* W3   = (const float*)d_in[5];
    const float* b3   = (const float*)d_in[6];
    float* out = (float*)d_out;

    // ---- workspace layout (256B aligned) ----
    char* ws = (char*)d_ws;
    size_t off = 0;
    auto alloc = [&](size_t bytes) -> char* {
        char* p = ws + off;
        off = (off + bytes + 255) & ~(size_t)255;
        return p;
    };
    unsigned short* X    = (unsigned short*)alloc((size_t)T_STEPS * B_ROWS * L0 * 2);
    unsigned short* W1b  = (unsigned short*)alloc((size_t)L1 * L0 * 2);
    unsigned short* W2b  = (unsigned short*)alloc((size_t)L2 * L1 * 2);
    unsigned short* W3b  = (unsigned short*)alloc((size_t)L3 * L2 * 2);
    float*          mem1 = (float*)alloc((size_t)B_ROWS * L1 * 4);
    float*          mem2 = (float*)alloc((size_t)B_ROWS * L2 * 4);
    float*          mem3 = (float*)alloc((size_t)B_ROWS * L3 * 4);
    unsigned short* spk1 = (unsigned short*)alloc((size_t)B_ROWS * L1 * 2);
    unsigned short* spk2 = (unsigned short*)alloc((size_t)B_ROWS * L2 * 2);
    unsigned short* spk3 = (unsigned short*)alloc((size_t)B_ROWS * L3 * 2);
    (void)ws_size; (void)in_sizes; (void)n_in; (void)out_size;

    const int TB = 256;
    auto cdiv = [](int a, int b) { return (a + b - 1) / b; };

    // ---- one-time per call: weights -> bf16, spikes, zero membranes ----
    cvt_bf16_kernel<<<cdiv(L1 * L0, TB), TB, 0, stream>>>(W1, W1b, L1 * L0);
    cvt_bf16_kernel<<<cdiv(L2 * L1, TB), TB, 0, stream>>>(W2, W2b, L2 * L1);
    cvt_bf16_kernel<<<cdiv(L3 * L2, TB), TB, 0, stream>>>(W3, W3b, L3 * L2);

    const int totX = T_STEPS * B_ROWS * L0;
    spikegen_kernel<<<cdiv(totX, TB), TB, 0, stream>>>(data, X, totX);

    zero_f32_kernel<<<cdiv(B_ROWS * L1, TB), TB, 0, stream>>>(mem1, B_ROWS * L1);
    zero_f32_kernel<<<cdiv(B_ROWS * L2, TB), TB, 0, stream>>>(mem2, B_ROWS * L2);
    zero_f32_kernel<<<cdiv(B_ROWS * L3, TB), TB, 0, stream>>>(mem3, B_ROWS * L3);

    // ---- T timesteps, 3 fused GEMM+LIF layers each ----
    for (int t = 0; t < T_STEPS; ++t) {
        const unsigned short* xt = X + (size_t)t * B_ROWS * L0;

        // layer 1: (256 x 2048) @ (4096 x 2048)^T
        {
            dim3 grid(L1 / BN, B_ROWS / BM);
            gemm_lif_bf16_wmma_kernel<<<grid, 256, 0, stream>>>(
                xt, W1b, b1, mem1, spk1, nullptr, B_ROWS, L1, L0);
        }
        // layer 2: (256 x 4096) @ (4096 x 4096)^T
        {
            dim3 grid(L2 / BN, B_ROWS / BM);
            gemm_lif_bf16_wmma_kernel<<<grid, 256, 0, stream>>>(
                spk1, W2b, b2, mem2, spk2, nullptr, B_ROWS, L2, L1);
        }
        // layer 3: (256 x 4096) @ (1024 x 4096)^T, spikes recorded to output
        {
            dim3 grid(L3 / BN, B_ROWS / BM);
            gemm_lif_bf16_wmma_kernel<<<grid, 256, 0, stream>>>(
                spk2, W3b, b3, mem3, spk3, out + (size_t)t * B_ROWS * L3,
                B_ROWS, L3, L2);
        }
    }
}
